// ELM_35038343201628
// MI455X (gfx1250) — matrix operand order repaired and checked by reading, not compiled, not run on hardware
//
#include <hip/hip_runtime.h>
#include <hip/hip_bf16.h>

// ---------------------------------------------------------------------------
// ELM recurrent cell, persistent-kernel formulation for MI455X (gfx1250).
//   B=128, T=1024, S=1000 synapses, M=100 memories, H=200 hidden.
//   8 workgroups x 16 batch rows; state on-chip for the whole T loop.
//   GEMMs via v_wmma_f32_16x16x32_bf16 (wave32 WMMA); W1 streamed from L2
//   one chunk ahead; readout y(t-1) runs on wave 13 concurrently with GEMM1.
// ---------------------------------------------------------------------------

typedef __attribute__((ext_vector_type(16))) __bf16 bf16x16;
typedef __attribute__((ext_vector_type(8)))  __bf16 bf16x8;
typedef __attribute__((ext_vector_type(8)))  float  f32x8;

union BF16x16 { bf16x16 v; bf16x8 h[2]; };

#define ELM_S    1000
#define ELM_M    100
#define ELM_H    200
#define ELM_T    1024
#define ELM_BB   16          // batch rows per workgroup (WMMA M)
#define ELM_NTHR 448         // 14 waves
#define KP1      1120        // padded K for GEMM1 (35 chunks of 32)
#define KC1      35
#define NT1      13          // N tiles of 16 (208 >= 200)
#define HP       208         // padded H
#define KP2      224         // padded K for GEMM2 (7 chunks of 32)
#define KC2      7
#define NT2      7           // 112 >= 100
#define MP       112         // padded M

// --- pack W1 [1100,200] f32 -> bf16 blocks [kc][n][512] in B-operand layout.
// Within a 32x16 block: lane ln holds col n=ln&15, K = (ln>>4)*16 + j, j=0..15,
// stored contiguously so each lane loads 32 bytes (2x b128).
__global__ void ELM_pack_w1(const float* __restrict__ W1, __bf16* __restrict__ w1p) {
  int idx = blockIdx.x * 256 + threadIdx.x;
  if (idx >= KC1 * NT1 * 512) return;
  int blk = idx >> 9, r = idx & 511;
  int kc = blk / NT1, n = blk % NT1;
  int ln = r >> 4, j = r & 15;
  int k = kc * 32 + (ln >> 4) * 16 + j;
  int c = n * 16 + (ln & 15);
  float v = (k < ELM_S + ELM_M && c < ELM_H) ? W1[k * ELM_H + c] : 0.0f;
  w1p[idx] = (__bf16)v;
}

// --- pack W2 [200,100] f32 -> bf16 blocks [kc][n][512], same layout.
__global__ void ELM_pack_w2(const float* __restrict__ W2, __bf16* __restrict__ w2p) {
  int idx = blockIdx.x * 256 + threadIdx.x;
  if (idx >= KC2 * NT2 * 512) return;
  int blk = idx >> 9, r = idx & 511;
  int kc = blk / NT2, n = blk % NT2;
  int ln = r >> 4, j = r & 15;
  int k = kc * 32 + (ln >> 4) * 16 + j;
  int c = n * 16 + (ln & 15);
  float v = (k < ELM_H && c < ELM_M) ? W2[k * ELM_M + c] : 0.0f;
  w2p[idx] = (__bf16)v;
}

__global__ __launch_bounds__(ELM_NTHR, 1) void ELM_persistent(
    const float*  __restrict__ X,     // [128,1024,1000]
    const float*  __restrict__ w_s,   // [1000]
    const float*  __restrict__ b1,    // [200]
    const float*  __restrict__ b2,    // [100]
    const float*  __restrict__ Wy,    // [100,10]
    const float*  __restrict__ by,    // [10]
    const __bf16* __restrict__ w1p,   // packed [35][13][512]
    const __bf16* __restrict__ w2p,   // packed [7][7][512]
    float*        __restrict__ out)   // [128,1024,10]
{
  __shared__ __bf16 inpA[ELM_BB * KP1];   // A operand for GEMM1 (s | km | pad)
  __shared__ __bf16 hA  [ELM_BB * KP2];   // A operand for GEMM2
  __shared__ float  mBuf[ELM_BB * MP];    // memory state m (cols 100..111 = 0)
  __shared__ float  wspos [ELM_S];
  __shared__ float  kappaM[MP];           // zero-padded to 112
  __shared__ float  b1s[HP];              // zero-padded to 208
  __shared__ float  b2s[MP];              // zero-padded to 112
  __shared__ float  WyS[ELM_M * 10];
  __shared__ float  byS[10];

  const int tid   = threadIdx.x;
  const int wave  = tid >> 5;
  const int lane  = tid & 31;
  const int wg    = blockIdx.x;      // 0..7, 16 batch rows each
  const int mrow  = lane & 15;       // WMMA row (A) / col (B,C)
  const int khalf = lane >> 4;

  // ---- one-time init ----
  for (int i = tid; i < ELM_S; i += ELM_NTHR) wspos[i] = fmaxf(w_s[i], 0.0f);
  {
    const float l0 = log10f(10.0f + 1e-6f);
    const float l1 = log10f(1000.0f - 1e-6f);
    for (int i = tid; i < MP; i += ELM_NTHR) {
      float kp = 0.0f;
      if (i < ELM_M) {
        float tau = powf(10.0f, l0 + (l1 - l0) * ((float)i / 99.0f));
        kp = expf(-1.0f / tau);
      }
      kappaM[i] = kp;
      b2s[i]    = (i < ELM_M) ? b2[i] : 0.0f;
    }
  }
  for (int i = tid; i < HP; i += ELM_NTHR)          b1s[i] = (i < ELM_H) ? b1[i] : 0.0f;
  for (int i = tid; i < ELM_M * 10; i += ELM_NTHR)  WyS[i] = Wy[i];
  if (tid < 10)                                     byS[tid] = by[tid];
  for (int i = tid; i < ELM_BB * MP; i += ELM_NTHR) mBuf[i] = 0.0f;
  // static zero padding of A operands (never rewritten)
  for (int i = tid; i < ELM_BB * 20; i += ELM_NTHR)
    inpA[(i / 20) * KP1 + (ELM_S + ELM_M) + (i % 20)] = (__bf16)0.0f;
  for (int i = tid; i < ELM_BB * 16; i += ELM_NTHR)
    hA[(i / 16) * KP2 + HP + (i % 16)] = (__bf16)0.0f;

  // synapse state lives in registers: 16000 elems over 448 threads -> 36 each
  float sreg[36];
  #pragma unroll
  for (int i = 0; i < 36; i++) sreg[i] = 0.0f;

  // W2 B-operand tiles resident in registers of waves 0..6 (56 VGPRs/lane)
  BF16x16 b2t[KC2];
  if (wave < NT2) {
    #pragma unroll
    for (int kc = 0; kc < KC2; kc++) {
      const __bf16* bp = w2p + ((size_t)(kc * NT2 + wave)) * 512 + lane * 16;
      b2t[kc].h[0] = *(const bf16x8*)(bp);
      b2t[kc].h[1] = *(const bf16x8*)(bp + 8);
    }
  }

  const float kappaS = expf(-1.0f / 5.0f);
  __syncthreads();

  for (int t = 0; t < ELM_T; t++) {
    // ---- prefetch next timestep's X slice (one lane per 128B line) ----
    if (t + 1 < ELM_T) {
      for (int i = tid; i < ELM_BB * 32; i += ELM_NTHR) {
        int row = i >> 5, off = (i & 31) * 32;   // 32 floats per 128B line
        if (off < ELM_S)
          __builtin_prefetch(
              X + ((size_t)(wg * ELM_BB + row) * ELM_T + (t + 1)) * ELM_S + off, 0, 3);
      }
    }

    // ---- phase A: s = ks*s + relu(w_s)*x_t ; km = kappa_m*m; build bf16 A
    #pragma unroll
    for (int i = 0; i < 36; i++) {
      int e = tid + i * ELM_NTHR;
      if (e < ELM_BB * ELM_S) {
        int row = e / ELM_S;
        int syn = e - row * ELM_S;
        size_t gx = ((size_t)(wg * ELM_BB + row) * ELM_T + t) * ELM_S + syn;
        float sv = kappaS * sreg[i] + wspos[syn] * X[gx];
        sreg[i] = sv;
        inpA[row * KP1 + syn] = (__bf16)sv;
      }
    }
    for (int i = tid; i < ELM_BB * ELM_M; i += ELM_NTHR) {
      int row = i / ELM_M, j = i - row * ELM_M;
      float km = kappaM[j] * mBuf[row * MP + j];
      inpA[row * KP1 + ELM_S + j] = (__bf16)km;
    }
    __syncthreads();

    // ---- GEMM1 (waves 0..12) + overlapped readout of y(t-1) (wave 13) ----
    if (wave < NT1) {
      f32x8 acc = {};
      BF16x16 bcur;
      {
        const __bf16* bp = w1p + ((size_t)wave) * 512 + lane * 16;
        bcur.h[0] = *(const bf16x8*)(bp);
        bcur.h[1] = *(const bf16x8*)(bp + 8);
      }
      for (int kc = 0; kc < KC1; kc++) {
        BF16x16 bnext = bcur;
        if (kc + 1 < KC1) {   // stream next W1 tile from L2 while MACing
          const __bf16* bp = w1p + ((size_t)((kc + 1) * NT1 + wave)) * 512 + lane * 16;
          bnext.h[0] = *(const bf16x8*)(bp);
          bnext.h[1] = *(const bf16x8*)(bp + 8);
        }
        BF16x16 a;
        const __bf16* ap = &inpA[mrow * KP1 + kc * 32 + khalf * 16];
        a.h[0] = *(const bf16x8*)(ap);
        a.h[1] = *(const bf16x8*)(ap + 8);
        acc = __builtin_amdgcn_wmma_f32_16x16x32_bf16(
            false, a.v, false, bcur.v, (short)0, acc, false, false);
        bcur = bnext;
      }
      // branchless epilogue: padded cols have zero weights + zero bias -> 0
      int Nc = wave * 16 + mrow;
      #pragma unroll
      for (int r = 0; r < 8; r++) {
        int Mr = r + khalf * 8;
        float hv = fmaxf(acc[r] + b1s[Nc], 0.0f);
        hA[Mr * KP2 + Nc] = (__bf16)hv;
      }
    } else if (wave == 13 && t > 0) {
      // y(t-1) = m(t-1) @ Wy + by; mBuf is stable until the next barrier + GEMM2
      #pragma unroll
      for (int q = 0; q < 5; q++) {
        int p = lane + 32 * q;                   // 160 (row, o) pairs
        int row = p / 10, o = p - row * 10;
        float acc = byS[o];
        #pragma unroll 4
        for (int j = 0; j < ELM_M; j++) acc += mBuf[row * MP + j] * WyS[j * 10 + o];
        out[((size_t)(wg * ELM_BB + row) * ELM_T + (t - 1)) * 10 + o] = acc;
      }
    }
    __syncthreads();

    // ---- GEMM2: dm = 1.7159*tanh((h@W2 + b2)*2/3); m = km + 3(1-k)dm ----
    if (wave < NT2) {
      f32x8 acc = {};
      #pragma unroll
      for (int kc = 0; kc < KC2; kc++) {
        BF16x16 a;
        const __bf16* ap = &hA[mrow * KP2 + kc * 32 + khalf * 16];
        a.h[0] = *(const bf16x8*)(ap);
        a.h[1] = *(const bf16x8*)(ap + 8);
        acc = __builtin_amdgcn_wmma_f32_16x16x32_bf16(
            false, a.v, false, b2t[kc].v, (short)0, acc, false, false);
      }
      // branchless epilogue: pad cols -> kp=0, b2s=0, acc=0  =>  m stays 0
      int Nc = wave * 16 + mrow;
      #pragma unroll
      for (int r = 0; r < 8; r++) {
        int Mr = r + khalf * 8;
        float u  = acc[r] + b2s[Nc];
        float dm = 1.7159f * tanhf(u * (2.0f / 3.0f));
        float kp = kappaM[Nc];
        int   mi = Mr * MP + Nc;
        float km = kp * mBuf[mi];
        mBuf[mi] = km + 3.0f * (1.0f - kp) * dm;
      }
    }
    __syncthreads();
    // next iteration's phase A / wave-13 readout only read mBuf; GEMM2's next
    // writes are ordered behind the two barriers above.
  }

  // ---- final readout flush: y(T-1) ----
  if (tid < ELM_BB * 10) {
    int row = tid / 10, o = tid - row * 10;
    float acc = byS[o];
    #pragma unroll 4
    for (int j = 0; j < ELM_M; j++) acc += mBuf[row * MP + j] * WyS[j * 10 + o];
    out[((size_t)(wg * ELM_BB + row) * ELM_T + (ELM_T - 1)) * 10 + o] = acc;
  }
}

extern "C" void kernel_launch(void* const* d_in, const int* in_sizes, int n_in,
                              void* d_out, int out_size, void* d_ws, size_t ws_size,
                              hipStream_t stream) {
  const float* X   = (const float*)d_in[0];
  const float* w_s = (const float*)d_in[1];
  const float* W1  = (const float*)d_in[2];
  const float* b1  = (const float*)d_in[3];
  const float* W2  = (const float*)d_in[4];
  const float* b2  = (const float*)d_in[5];
  const float* Wy  = (const float*)d_in[6];
  const float* by  = (const float*)d_in[7];
  float* out = (float*)d_out;

  // workspace layout: packed bf16 weights (W1p 465,920 B; W2p 50,176 B)
  __bf16* w1p = (__bf16*)d_ws;
  __bf16* w2p = (__bf16*)((char*)d_ws + 466944);

  ELM_pack_w1<<<(KC1 * NT1 * 512 + 255) / 256, 256, 0, stream>>>(W1, w1p);
  ELM_pack_w2<<<(KC2 * NT2 * 512 + 255) / 256, 256, 0, stream>>>(W2, w2p);
  ELM_persistent<<<8, ELM_NTHR, 0, stream>>>(X, w_s, b1, b2, Wy, by, w1p, w2p, out);
}
